// FCOS_17197049053809
// MI455X (gfx1250) — matrix operand order repaired and checked by reading, not compiled, not run on hardware
//
#include <hip/hip_runtime.h>
#include <cstdint>
#include <cstddef>

// ---------------------------------------------------------------------------
// Types for CDNA5 WMMA (wave32): V_WMMA_F32_16X16X32_BF16
// ---------------------------------------------------------------------------
typedef __bf16 bf16_t;
typedef __attribute__((ext_vector_type(16))) __bf16 v16bf;
typedef __attribute__((ext_vector_type(8)))  __bf16 v8bf;
typedef __attribute__((ext_vector_type(8)))  float  v8f;
typedef __attribute__((ext_vector_type(4)))  int    v4i;

union AFrag { v16bf v; v8bf h[2]; };
struct A2 { AFrag f0, f1; };

// The async builtin takes int4* in global (AS1) / LDS (AS3) address spaces.
typedef __attribute__((address_space(1))) v4i gas_v4i;
typedef __attribute__((address_space(3))) v4i las_v4i;

#if __has_builtin(__builtin_amdgcn_global_load_async_to_lds_b128)
#define HAS_ASYNC_LDS 1
#else
#define HAS_ASYNC_LDS 0
#endif

__device__ __forceinline__ void wait_asynccnt_le8() {
#if __has_builtin(__builtin_amdgcn_s_wait_asynccnt)
  __builtin_amdgcn_s_wait_asynccnt(8);
#elif HAS_ASYNC_LDS
  asm volatile("s_wait_asynccnt 0x8" ::: "memory");
#endif
}
__device__ __forceinline__ void wait_asynccnt_0() {
#if __has_builtin(__builtin_amdgcn_s_wait_asynccnt)
  __builtin_amdgcn_s_wait_asynccnt(0);
#elif HAS_ASYNC_LDS
  asm volatile("s_wait_asynccnt 0x0" ::: "memory");
#endif
}

// Leading barrier: pure execution barrier (producers already waited their
// ASYNCcnt), no counter flush -> no stall in the latency-critical path.
__device__ __forceinline__ void wg_barrier_publish() {
  asm volatile("s_barrier_signal -1\n\ts_barrier_wait -1" ::: "memory");
}
// Trailing barrier: must drain this wave's outstanding LDS reads before
// signaling, because the next chunk's async DMA (issued right after other
// waves pass this barrier) overwrites the buffer parity we just read.
__device__ __forceinline__ void wg_barrier_drain_reads() {
  asm volatile("s_wait_dscnt 0x0\n\ts_barrier_signal -1\n\ts_barrier_wait -1"
               ::: "memory");
}

// LDS B-tile geometry: rows = out-channels, 64 K-values (128B) + 16B pad.
// Pitch 144B -> bank index 36*co mod 64, coprime stride => conflict-free.
#define BPITCH 72   // elements (bf16) per row

// ---------------------------------------------------------------------------
// Stage one 64-wide K-chunk of B ([NCO][64] bf16) into LDS.
// Async DMA (ASYNCcnt) when available; 8 b128 transfers per row.
// ---------------------------------------------------------------------------
template <int NCO>
__device__ __forceinline__ void stage_b_tile(const bf16_t* __restrict__ wt,
                                             int kk, bf16_t* lrow0) {
  for (int co = threadIdx.x; co < NCO; co += 256) {
    const bf16_t* g = wt + (size_t)co * 2304 + kk;  // 128 contiguous bytes
    bf16_t* l = lrow0 + co * BPITCH;
#if HAS_ASYNC_LDS
#pragma unroll
    for (int j = 0; j < 8; ++j)
      __builtin_amdgcn_global_load_async_to_lds_b128(
          (gas_v4i*)(g + j * 8), (las_v4i*)(l + j * 8), 0, 0);
#else
#pragma unroll
    for (int j = 0; j < 8; ++j) {
      v8bf tv = *(const v8bf*)(g + j * 8);
      *(v8bf*)(l + j * 8) = tv;
    }
#endif
  }
}

// ---------------------------------------------------------------------------
// Load the two A fragments (K-substeps) of chunk c for this lane's row.
// ---------------------------------------------------------------------------
__device__ __forceinline__ A2 load_a_chunk(const bf16_t* __restrict__ act,
                                           int n, int y, int x, bool rowok,
                                           int H, int W, int HW, int c,
                                           int kh) {
  const int tap = c >> 2;
  const int cb = (c & 3) * 64;
  const int ky = tap / 3;
  const int kx = tap - 3 * ky;
  const int yy = y + ky - 1;
  const int xx = x + kx - 1;
  const bool pixok = rowok && (yy >= 0) && (yy < H) && (xx >= 0) && (xx < W);
  A2 r;
  union { v8bf v; unsigned u[4]; } z;
  z.u[0] = z.u[1] = z.u[2] = z.u[3] = 0u;
  r.f0.h[0] = r.f0.h[1] = r.f1.h[0] = r.f1.h[1] = z.v;   // zero padding
  if (pixok) {
    const bf16_t* p = act + (((size_t)(n * HW + yy * W + xx)) << 8) + cb + kh * 16;
    r.f0.h[0] = *(const v8bf*)(p);        // K = cb + 16*kh + [0..7]
    r.f0.h[1] = *(const v8bf*)(p + 8);    // K = cb + 16*kh + [8..15]
    r.f1.h[0] = *(const v8bf*)(p + 32);   // substep 1: K = cb+32 + ...
    r.f1.h[1] = *(const v8bf*)(p + 40);
  }
  return r;
}

// ---------------------------------------------------------------------------
// Weight transform: OIHW fp32 -> [co][tap*256+ci] bf16 (K-major rows).
// ---------------------------------------------------------------------------
__global__ void wxform_kernel(const float* __restrict__ s1,
                              const float* __restrict__ s2,
                              int n1, int n2,
                              bf16_t* __restrict__ dst, int CO) {
  int idx = blockIdx.x * blockDim.x + threadIdx.x;
  if (idx >= CO * 2304) return;
  int co = idx / 2304;
  int k = idx - co * 2304;
  int tap = k >> 8;
  int ci = k & 255;
  float v = 0.f;
  if (co < n1)           v = s1[((size_t)(co * 256 + ci)) * 9 + tap];
  else if (co < n1 + n2) v = s2[((size_t)((co - n1) * 256 + ci)) * 9 + tap];
  dst[idx] = (bf16_t)v;
}

__global__ void biaspad_kernel(const float* __restrict__ s1,
                               const float* __restrict__ s2,
                               int n1, int n2,
                               float* __restrict__ dst, int CO) {
  int i = threadIdx.x;
  if (i >= CO) return;
  float v = 0.f;
  if (i < n1)           v = s1[i];
  else if (i < n1 + n2) v = s2[i - n1];
  dst[i] = v;
}

__global__ void nchw_to_nhwc_kernel(const float* __restrict__ in,
                                    bf16_t* __restrict__ outp, int HW) {
  int idx = blockIdx.x * blockDim.x + threadIdx.x;
  if (idx >= 2 * 256 * HW) return;
  int m = idx >> 8;
  int c = idx & 255;
  int nn = m / HW;
  int p = m - nn * HW;
  outp[idx] = (bf16_t)in[((size_t)(nn * 256 + c)) * HW + p];
}

// ---------------------------------------------------------------------------
// Implicit-GEMM 3x3 SAME conv via WMMA bf16->f32, double-buffered pipeline:
//   - B tile (weights) staged to LDS via async DMA, chunk K=64, ping-pong
//   - A fragments register double-buffered, prefetched under the WMMAs
//   - B fragments batch-loaded from LDS (4 at a time) -> staggered dscnt waits
// MODE 0: tower layer  -> conv+bias, fp32 NHWC to convout
// MODE 1: cls head     -> conv+bias, write d_out channels [5..84]
// MODE 2: reg+iou head -> exp/scale + FCOS grid decode, d_out channels [0..4]
// ---------------------------------------------------------------------------
template <int NT, int MODE>
__global__ __launch_bounds__(256) void conv3x3_wmma_kernel(
    const bf16_t* __restrict__ act,    // [2*H*W][256] bf16 NHWC
    const bf16_t* __restrict__ wt,     // [NT*16][2304] bf16
    const float*  __restrict__ bias,   // [NT*16]
    float* __restrict__ convout,       // MODE 0
    float* __restrict__ out,           // MODE 1/2: d_out base
    const float* __restrict__ scales,  // MODE 2
    int H, int W, int level_off, int level, float stride_px) {
  constexpr int TILES = 4 * NT;
  constexpr int PW = (TILES + 7) / 8;  // N-subtiles per wave
  constexpr int NCO = NT * 16;
  constexpr int NCHUNK = 36;           // 2304 / 64

  __shared__ __align__(64) bf16_t bsm[2][256 * BPITCH];

  const int HW = H * W;
  const int Mtot = 2 * HW;
  // Wave id as an SGPR (scalar branches around WMMA) AND with a provable
  // range [0,7] so nbase in [0,1] lets the per-tile guards constant-fold.
  const int wave = __builtin_amdgcn_readfirstlane((int)(threadIdx.x >> 5)) & 7;
  const int lane = threadIdx.x & 31;
  const int m0 = blockIdx.x * 64;
  const int msub = wave & 3;           // [0,3]
  const int nbase = wave >> 2;         // [0,1] -- provable
  const int colp = lane & 15;
  const int kh = lane >> 4;
  const bool wactive = (nbase < NT);   // folds to true for NT >= 2

  v8f acc[PW];
#pragma unroll
  for (int t = 0; t < PW; ++t)
#pragma unroll
    for (int e = 0; e < 8; ++e) acc[t][e] = 0.f;

  // A-fragment row geometry (lanes 0-15 / 16-31 hold the same 16 rows).
  const int mrow = m0 + msub * 16 + colp;
  const bool rowok = mrow < Mtot;
  int n = 0, y = 0, x = 0;
  if (rowok) {
    n = mrow / HW;
    int rem = mrow - n * HW;
    y = rem / W;
    x = rem - y * W;
  }

  // ---- Pipelined mainloop ----
  stage_b_tile<NCO>(wt, 0, &bsm[0][0]);
  A2 a_cur;
  if (wactive) a_cur = load_a_chunk(act, n, y, x, rowok, H, W, HW, 0, kh);

  for (int c = 0; c < NCHUNK; ++c) {
    const int buf = c & 1;
    const bool hasnext = (c + 1) < NCHUNK;
    if (hasnext)
      stage_b_tile<NCO>(wt, (c + 1) * 64, &bsm[buf ^ 1][0]);   // ASYNCcnt
    if (hasnext) wait_asynccnt_le8();   // current buffer's DMA complete
    else         wait_asynccnt_0();
    wg_barrier_publish();               // publish LDS tile (no counter flush)
    // A prefetch issued here overlaps the 16 WMMAs below; its completion is
    // only required at the a_cur = a_nxt copy at the end of the chunk.
    A2 a_nxt;
    if (hasnext && wactive)
      a_nxt = load_a_chunk(act, n, y, x, rowok, H, W, HW, c + 1, kh);
    if (wactive) {
#pragma unroll
      for (int s = 0; s < 2; ++s) {
        const v16bf afv = (s == 0) ? a_cur.f0.v : a_cur.f1.v;
#pragma unroll
        for (int g0 = 0; g0 < PW; g0 += 4) {
          v16bf bfr[4];
#pragma unroll
          for (int u = 0; u < 4; ++u) {
            const int t = g0 + u;
            if (t < PW) {
              const int nsub = nbase + 2 * t;
              if (nsub < NT)
                bfr[u] = *(const v16bf*)&bsm[buf][(nsub * 16 + colp) * BPITCH +
                                                  s * 32 + kh * 16];
            }
          }
#pragma unroll
          for (int u = 0; u < 4; ++u) {
            const int t = g0 + u;
            if (t < PW) {
              const int nsub = nbase + 2 * t;
              if (nsub < NT)
                acc[t] = __builtin_amdgcn_wmma_f32_16x16x32_bf16(
                    false, afv, false, bfr[u], (short)0, acc[t], false, false);
            }
          }
        }
      }
      a_cur = a_nxt;
    }
    wg_barrier_drain_reads();           // LDS reads complete, then barrier
  }

  // ---- Epilogues (C/D layout: lane col = colp, row = kh*8 + v) ----
  if constexpr (MODE == 0) {
#pragma unroll
    for (int t = 0; t < PW; ++t) {
      const int nsub = nbase + 2 * t;
      if (nsub < NT) {
        const int co = nsub * 16 + colp;
        const float bv = bias[co];
#pragma unroll
        for (int v = 0; v < 8; ++v) {
          const int mr = m0 + msub * 16 + kh * 8 + v;
          if (mr < Mtot) convout[(size_t)mr * 256 + co] = acc[t][v] + bv;
        }
      }
    }
  }
  if constexpr (MODE == 1) {
#pragma unroll
    for (int t = 0; t < PW; ++t) {
      const int nsub = nbase + 2 * t;
      if (nsub < NT) {
        const int co = nsub * 16 + colp;
        if (co < 80) {
          const float bv = bias[co];
#pragma unroll
          for (int v = 0; v < 8; ++v) {
            const int mr = m0 + msub * 16 + kh * 8 + v;
            if (mr < Mtot) {
              const int nn = (mr >= HW) ? 1 : 0;
              const int pix = mr - nn * HW;
              out[((size_t)(nn * 13343 + level_off + pix)) * 85 + 5 + co] =
                  acc[t][v] + bv;
            }
          }
        }
      }
    }
  }
  if constexpr (MODE == 2) {
    if (wactive) {
      const float sc = scales[level];
      const int co = colp;              // packed: 0-3 reg, 4 iou
      if (co < 5) {
        const float bv = bias[co];
#pragma unroll
        for (int v = 0; v < 8; ++v) {
          const int mr = m0 + msub * 16 + kh * 8 + v;
          if (mr < Mtot) {
            const int nn = (mr >= HW) ? 1 : 0;
            const int pix = mr - nn * HW;
            const float a = acc[0][v] + bv;
            float r;
            if (co == 4) {
              r = a;                    // centerness, raw
            } else {
              const int yp = pix / W;
              const int xp = pix - yp * W;
              const float e = __expf(a * sc);
              const float gx = (xp + 0.5f) * stride_px;
              const float gy = (yp + 0.5f) * stride_px;
              r = (co == 0) ? gx - e : (co == 1) ? gy - e
                                     : (co == 2) ? gx + e : gy + e;
            }
            out[((size_t)(nn * 13343 + level_off + pix)) * 85 + co] = r;
          }
        }
      }
    }
  }
}

// ---------------------------------------------------------------------------
// GroupNorm pass 1: deterministic sum/sumsq per (image, group), LDS tree.
// ---------------------------------------------------------------------------
__global__ __launch_bounds__(256) void gn_stats_kernel(
    const float* __restrict__ convout, float* __restrict__ stats, int HW) {
  const int nn = blockIdx.x >> 5;
  const int g = blockIdx.x & 31;
  const int tid = threadIdx.x;
  const int total = 8 * HW;
  float s = 0.f, q = 0.f;
  for (int j = tid; j < total; j += 256) {
    const int ch = (g << 3) + (j & 7);
    const int pix = j >> 3;
    const float v = convout[((size_t)(nn * HW + pix)) * 256 + ch];
    s += v;
    q += v * v;
  }
  __shared__ float sh_s[256];
  __shared__ float sh_q[256];
  sh_s[tid] = s;
  sh_q[tid] = q;
  __syncthreads();
  for (int off = 128; off > 0; off >>= 1) {
    if (tid < off) {
      sh_s[tid] += sh_s[tid + off];
      sh_q[tid] += sh_q[tid + off];
    }
    __syncthreads();
  }
  if (tid == 0) {
    stats[(nn * 32 + g) * 2 + 0] = sh_s[0];
    stats[(nn * 32 + g) * 2 + 1] = sh_q[0];
  }
}

// ---------------------------------------------------------------------------
// GroupNorm pass 2 + affine + ReLU + bf16 convert.
// ---------------------------------------------------------------------------
__global__ void gn_relu_kernel(const float* __restrict__ convout,
                               const float* __restrict__ stats,
                               const float* __restrict__ gamma,
                               const float* __restrict__ beta,
                               bf16_t* __restrict__ actout, int HW) {
  int idx = blockIdx.x * blockDim.x + threadIdx.x;
  const int Mtot = 2 * HW;
  if (idx >= Mtot * 256) return;
  const int m = idx >> 8;
  const int c = idx & 255;
  const int nn = (m >= HW) ? 1 : 0;
  const int g = c >> 3;
  const float cnt = 8.f * (float)HW;
  const float s = stats[(nn * 32 + g) * 2 + 0];
  const float q = stats[(nn * 32 + g) * 2 + 1];
  const float mean = s / cnt;
  const float var = q / cnt - mean * mean;
  const float rstd = rsqrtf(var + 1e-5f);
  float v = (convout[idx] - mean) * rstd * gamma[c] + beta[c];
  v = fmaxf(v, 0.f);
  actout[idx] = (bf16_t)v;
}

// ---------------------------------------------------------------------------
// Host orchestration
// ---------------------------------------------------------------------------
extern "C" void kernel_launch(void* const* d_in, const int* in_sizes, int n_in,
                              void* d_out, int out_size, void* d_ws,
                              size_t ws_size, hipStream_t stream) {
  (void)in_sizes; (void)n_in; (void)out_size; (void)ws_size;
  const float* P[5];
  for (int i = 0; i < 5; ++i) P[i] = (const float*)d_in[i];
  const float* cls_tw   = (const float*)d_in[5];
  const float* cls_tb   = (const float*)d_in[6];
  const float* cls_gn_g = (const float*)d_in[7];
  const float* cls_gn_b = (const float*)d_in[8];
  const float* reg_tw   = (const float*)d_in[9];
  const float* reg_tb   = (const float*)d_in[10];
  const float* reg_gn_g = (const float*)d_in[11];
  const float* reg_gn_b = (const float*)d_in[12];
  const float* cls_hw   = (const float*)d_in[13];
  const float* cls_hb   = (const float*)d_in[14];
  const float* reg_hw   = (const float*)d_in[15];
  const float* reg_hb   = (const float*)d_in[16];
  const float* iou_hw   = (const float*)d_in[17];
  const float* iou_hb   = (const float*)d_in[18];
  const float* scales   = (const float*)d_in[19];
  float* outp = (float*)d_out;

  char* wsp = (char*)d_ws;
  auto carve = [&](size_t bytes) -> void* {
    void* r = (void*)wsp;
    wsp += (bytes + 255) & ~(size_t)255;
    return r;
  };
  const int MAXHW = 10000;
  const size_t ACT_B = (size_t)2 * MAXHW * 256 * sizeof(bf16_t);
  bf16_t* act0 = (bf16_t*)carve(ACT_B);
  bf16_t* actA = (bf16_t*)carve(ACT_B);
  bf16_t* actB = (bf16_t*)carve(ACT_B);
  bf16_t* actC = (bf16_t*)carve(ACT_B);
  float* convtmp = (float*)carve((size_t)2 * MAXHW * 256 * sizeof(float));
  bf16_t* wbuf[8];
  for (int l = 0; l < 8; ++l)
    wbuf[l] = (bf16_t*)carve((size_t)256 * 2304 * sizeof(bf16_t));
  bf16_t* wcls = (bf16_t*)carve((size_t)96 * 2304 * sizeof(bf16_t));
  bf16_t* wri  = (bf16_t*)carve((size_t)16 * 2304 * sizeof(bf16_t));
  float* bcls = (float*)carve(96 * sizeof(float));
  float* bri  = (float*)carve(16 * sizeof(float));
  float* statsbuf = (float*)carve(128 * sizeof(float));  // [2][32][2]

  const size_t WSTEP = (size_t)256 * 256 * 9;
  const int wg256 = (256 * 2304 + 255) / 256;
  for (int l = 0; l < 4; ++l) {
    wxform_kernel<<<wg256, 256, 0, stream>>>(cls_tw + l * WSTEP, nullptr, 256,
                                             0, wbuf[l], 256);
    wxform_kernel<<<wg256, 256, 0, stream>>>(reg_tw + l * WSTEP, nullptr, 256,
                                             0, wbuf[4 + l], 256);
  }
  wxform_kernel<<<(96 * 2304 + 255) / 256, 256, 0, stream>>>(
      cls_hw, nullptr, 80, 0, wcls, 96);
  wxform_kernel<<<(16 * 2304 + 255) / 256, 256, 0, stream>>>(
      reg_hw, iou_hw, 4, 1, wri, 16);
  biaspad_kernel<<<1, 128, 0, stream>>>(cls_hb, nullptr, 80, 0, bcls, 96);
  biaspad_kernel<<<1, 128, 0, stream>>>(reg_hb, iou_hb, 4, 1, bri, 16);

  static const int HS[5] = {100, 50, 25, 13, 7};
  static const float STR[5] = {8.f, 16.f, 32.f, 64.f, 128.f};
  int level_off = 0;
  for (int i = 0; i < 5; ++i) {
    const int H = HS[i], W = HS[i], HW = H * W, Mtot = 2 * HW;
    const int convGrid = (Mtot + 63) / 64;
    const int elemGrid = (Mtot * 256 + 255) / 256;
    nchw_to_nhwc_kernel<<<elemGrid, 256, 0, stream>>>(P[i], act0, HW);

    // cls tower: act0 -> A -> B -> A -> B
    const bf16_t* cur = act0;
    bf16_t* ping[2] = {actA, actB};
    for (int l = 0; l < 4; ++l) {
      conv3x3_wmma_kernel<16, 0><<<convGrid, 256, 0, stream>>>(
          cur, wbuf[l], cls_tb + l * 256, convtmp, nullptr, nullptr, H, W, 0,
          0, 0.f);
      gn_stats_kernel<<<64, 256, 0, stream>>>(convtmp, statsbuf, HW);
      gn_relu_kernel<<<elemGrid, 256, 0, stream>>>(
          convtmp, statsbuf, cls_gn_g + l * 256, cls_gn_b + l * 256,
          ping[l & 1], HW);
      cur = ping[l & 1];
    }
    const bf16_t* clsT = cur;  // actB

    // reg tower: act0 -> C -> A -> C -> A (actB stays alive)
    cur = act0;
    bf16_t* pong[2] = {actC, actA};
    for (int l = 0; l < 4; ++l) {
      conv3x3_wmma_kernel<16, 0><<<convGrid, 256, 0, stream>>>(
          cur, wbuf[4 + l], reg_tb + l * 256, convtmp, nullptr, nullptr, H, W,
          0, 0, 0.f);
      gn_stats_kernel<<<64, 256, 0, stream>>>(convtmp, statsbuf, HW);
      gn_relu_kernel<<<elemGrid, 256, 0, stream>>>(
          convtmp, statsbuf, reg_gn_g + l * 256, reg_gn_b + l * 256,
          pong[l & 1], HW);
      cur = pong[l & 1];
    }
    const bf16_t* regT = cur;  // actA

    conv3x3_wmma_kernel<6, 1><<<convGrid, 256, 0, stream>>>(
        clsT, wcls, bcls, nullptr, outp, nullptr, H, W, level_off, i, 0.f);
    conv3x3_wmma_kernel<1, 2><<<convGrid, 256, 0, stream>>>(
        regT, wri, bri, nullptr, outp, scales, H, W, level_off, i, STR[i]);
    level_off += HW;
  }
}
